// S5SSM_32796370272330
// MI455X (gfx1250) — compile-verified, matrix operand order checked by reading, not run on hardware
//
#include <hip/hip_runtime.h>
#include <hip/hip_bf16.h>

// S5 SSM forward for MI455X (gfx1250, wave32, WMMA).
// BSZ=8, H=512, P=256, L=4096.
// Pipeline: discretize -> pack W1/W2 (bf16, WMMA-fragment layout) ->
//           GEMM1 (Bu = u @ W1, bf16 WMMA) -> wave32 shuffle scan ->
//           GEMM2 (y = xs @ W2) + D*u + exact GeLU epilogue.
//
// GEMM blocks: 256 threads = 8 waves, tile M=64 x N=128. The full 64x512
// bf16 A panel is staged in LDS once (row pad +16B -> conflict-free
// ds_load_b128 fragment reads), then 16 K-tiles x 4 M-subtiles = 64 WMMAs
// per wave with no barriers inside the K loop. B operand comes pre-packed
// in WMMA-fragment order (contiguous 32B per lane, L2-resident).

#define BSZ 8
#define Hd  512
#define Pd  256
#define Ld  4096
#define Md  (BSZ * Ld)      // 32768 GEMM rows
#define Nd  512             // 2P (re|im) == H
#define Kd  512             // H == 2P

#define APAD 8              // row padding in bf16 elems (16B) -> bank spread
#define AROW (Kd + APAD)    // 520

typedef __attribute__((ext_vector_type(16))) __bf16 v16bf;
typedef __attribute__((ext_vector_type(8)))  __bf16 v8bf;
typedef __attribute__((ext_vector_type(8)))  float  v8f;

// 16-bit A/B matrix fragment k-mapping (CDNA5 ISA 7.12.2):
// lanes 0-15: halfwords 0..7 -> K=0..7, 8..15 -> K=16..23
// lanes 16-31: halfwords 0..7 -> K=8..15, 8..15 -> K=24..31
__device__ __forceinline__ int kmap(int i, int laneHi) {
    return (i & 7) + ((i >> 3) << 4) + (laneHi << 3);
}

__device__ __forceinline__ unsigned pack_bf16x2(float a, float b) {
    union { __bf16 h[2]; unsigned u; } r;
    r.h[0] = (__bf16)a;
    r.h[1] = (__bf16)b;
    return r.u;                      // -> single v_cvt_pk_bf16_f32
}

__device__ __forceinline__ v16bf cat8(v8bf lo, v8bf hi) {
    return __builtin_shufflevector(lo, hi,
        0,1,2,3,4,5,6,7,8,9,10,11,12,13,14,15);
}

// ---------------------------------------------------------------------------
// Kernel 1: ZOH discretization. lam[0..255]=Re(Lambda_bar), lam[256..511]=Im.
// g = (Lambda_bar - 1) / Lambda  (complex), same layout.
// ---------------------------------------------------------------------------
__global__ void s5_discretize(const float* __restrict__ Lre,
                              const float* __restrict__ Lim,
                              const float* __restrict__ log_step,
                              float* __restrict__ lam,
                              float* __restrict__ g) {
    int p = threadIdx.x;
    if (p >= Pd) return;
    float step = __expf(log_step[p]);
    float ar = Lre[p] * step, ai = Lim[p] * step;
    float er = __expf(ar);
    float lr = er * __cosf(ai);
    float li = er * __sinf(ai);
    lam[p]       = lr;
    lam[p + Pd]  = li;
    float nr = lr - 1.0f, ni = li;
    float dr = Lre[p], di = Lim[p];
    float inv = 1.0f / (dr * dr + di * di);
    g[p]       = (nr * dr + ni * di) * inv;
    g[p + Pd]  = (ni * dr - nr * di) * inv;
}

// ---------------------------------------------------------------------------
// Kernel 2: pack W1[k=h][n] (n<256: Re(B_bar[p=n][h]); n>=256: Im) into
// WMMA B-fragment layout: frag f = ktile*(N/16)+ntile, 32 lanes x 16 bf16.
// ---------------------------------------------------------------------------
__global__ void s5_pack_w1(const float* __restrict__ Bm,   // (P,H,2)
                           const float* __restrict__ g,
                           __bf16* __restrict__ W1) {
    int tid    = blockIdx.x * blockDim.x + threadIdx.x;   // 16384 threads
    int f      = tid >> 5;
    int lane   = tid & 31;
    int ktile  = f >> 5;           // K/32 = 16 tiles
    int ntile  = f & 31;           // N/16 = 32 tiles
    int nloc   = lane & 15;
    int laneHi = lane >> 4;
    int n      = ntile * 16 + nloc;
    __bf16* dst = W1 + (size_t)f * 512 + lane * 16;
#pragma unroll
    for (int i = 0; i < 16; ++i) {
        int h = ktile * 32 + kmap(i, laneHi);
        float v;
        if (n < Pd) {
            int p = n;
            float br = Bm[((size_t)p * Hd + h) * 2 + 0];
            float bi = Bm[((size_t)p * Hd + h) * 2 + 1];
            v = g[p] * br - g[p + Pd] * bi;               // Re(g * B_tilde)
        } else {
            int p = n - Pd;
            float br = Bm[((size_t)p * Hd + h) * 2 + 0];
            float bi = Bm[((size_t)p * Hd + h) * 2 + 1];
            v = g[p] * bi + g[p + Pd] * br;               // Im(g * B_tilde)
        }
        dst[i] = (__bf16)v;
    }
}

// ---------------------------------------------------------------------------
// Kernel 3: pack W2[k=p2][n=h] = (p2<256 ? 2*Cr[h][p2] : -2*Ci[h][p2-256]).
// ---------------------------------------------------------------------------
__global__ void s5_pack_w2(const float* __restrict__ Cm,   // (H,P,2)
                           __bf16* __restrict__ W2) {
    int tid    = blockIdx.x * blockDim.x + threadIdx.x;
    int f      = tid >> 5;
    int lane   = tid & 31;
    int ktile  = f >> 5;
    int ntile  = f & 31;
    int nloc   = lane & 15;
    int laneHi = lane >> 4;
    int h      = ntile * 16 + nloc;
    __bf16* dst = W2 + (size_t)f * 512 + lane * 16;
#pragma unroll
    for (int i = 0; i < 16; ++i) {
        int p2 = ktile * 32 + kmap(i, laneHi);
        float v;
        if (p2 < Pd)
            v =  2.0f * Cm[((size_t)h * Pd + p2) * 2 + 0];
        else
            v = -2.0f * Cm[((size_t)h * Pd + (p2 - Pd)) * 2 + 1];
        dst[i] = (__bf16)v;
    }
}

// ---------------------------------------------------------------------------
// Kernel 4: GEMM1  Bu[m][n] = sum_k u[m][k] * W1[k][n]
// u[m=b*L+l][k=h] = in[b*H*L + h*L + l]  (strided in k, contiguous in m).
// ---------------------------------------------------------------------------
__global__ void s5_gemm1(const float* __restrict__ in,
                         const __bf16* __restrict__ W1,
                         float* __restrict__ bu) {
    __shared__ __bf16 At[64][AROW];     // 64x(512+8) bf16 = 65 KB
    int tid  = threadIdx.x;
    int wave = tid >> 5;
    int lane = tid & 31;
    int m0   = blockIdx.x * 64;
    int n0   = blockIdx.y * 128 + wave * 16;
    int ntile = n0 >> 4;
    int b  = m0 >> 12;                  // L = 4096, 4096 % 64 == 0
    int l0 = m0 & (Ld - 1);
    const float* ubase = in + (size_t)b * Hd * Ld;

    // stage full 64x512 A panel: pairs along k packed -> one ds_store_b32
    // tids 0..63 map to consecutive m (contiguous global 256B segments)
#pragma unroll 4
    for (int it = 0; it < 64; ++it) {
        int idx = tid + it * 256;
        int m  = idx & 63;
        int k2 = idx >> 6;              // 0..255 pair index
        float a = ubase[(size_t)(2 * k2)     * Ld + (l0 + m)];
        float c = ubase[(size_t)(2 * k2 + 1) * Ld + (l0 + m)];
        *(unsigned*)&At[m][2 * k2] = pack_bf16x2(a, c);
    }
    __syncthreads();

    int hi   = lane >> 4;
    int mrow = lane & 15;
    v8f acc[4] = {};
    for (int kt = 0; kt < Kd / 32; ++kt) {
        const v8bf* wp = (const v8bf*)(W1 + ((size_t)(kt * 32 + ntile)) * 512
                                          + lane * 16);
        v16bf bfrag = cat8(wp[0], wp[1]);
#pragma unroll
        for (int sub = 0; sub < 4; ++sub) {
            const v8bf* arow = (const v8bf*)(&At[sub * 16 + mrow][kt * 32]);
            v16bf afrag = cat8(arow[hi], arow[hi + 2]);
            acc[sub] = __builtin_amdgcn_wmma_f32_16x16x32_bf16(
                false, afrag, false, bfrag, (short)0, acc[sub], false, false);
        }
    }
    // D layout: VGPR r -> (m = r + 8*(lane>>4), n = lane&15); n-contig store
    int nloc = lane & 15;
#pragma unroll
    for (int sub = 0; sub < 4; ++sub)
#pragma unroll
        for (int r = 0; r < 8; ++r) {
            int m = m0 + sub * 16 + r + 8 * hi;
            bu[(size_t)m * Nd + n0 + nloc] = acc[sub][r];
        }
}

// ---------------------------------------------------------------------------
// Kernel 5: complex inclusive scan over L per (b,p):  x_l = lam*x_{l-1} + Bu_l
// One wave per chain; lane owns 128 steps. Pass1: chunk transform (A,b);
// Hillis-Steele wave scan via shfl_up; Pass2: replay with carry-in. In-place.
// ---------------------------------------------------------------------------
__global__ void s5_scan(const float* __restrict__ lam,
                        float* __restrict__ bu) {
    int b    = blockIdx.y;
    int p    = blockIdx.x * 8 + (threadIdx.x >> 5);
    int lane = threadIdx.x & 31;
    float lre = lam[p], lim = lam[p + Pd];
    size_t base = (size_t)b * Ld * Nd;
    int l0 = lane * 128;
    float Ar = 1.0f, Ai = 0.0f, xr = 0.0f, xi = 0.0f;
    for (int t = 0; t < 128; ++t) {
        size_t row = base + (size_t)(l0 + t) * Nd;
        float br = bu[row + p];
        float bi = bu[row + p + Pd];
        float nxr = lre * xr - lim * xi + br;
        float nxi = lre * xi + lim * xr + bi;
        xr = nxr; xi = nxi;
        float nAr = lre * Ar - lim * Ai;
        float nAi = lre * Ai + lim * Ar;
        Ar = nAr; Ai = nAi;
    }
#pragma unroll
    for (int off = 1; off < 32; off <<= 1) {
        float pAr = __shfl_up(Ar, off, 32);
        float pAi = __shfl_up(Ai, off, 32);
        float pxr = __shfl_up(xr, off, 32);
        float pxi = __shfl_up(xi, off, 32);
        if (lane >= off) {
            float cAr = Ar * pAr - Ai * pAi;
            float cAi = Ar * pAi + Ai * pAr;
            float cxr = Ar * pxr - Ai * pxi + xr;
            float cxi = Ar * pxi + Ai * pxr + xi;
            Ar = cAr; Ai = cAi; xr = cxr; xi = cxi;
        }
    }
    float cr = __shfl_up(xr, 1, 32);
    float ci = __shfl_up(xi, 1, 32);
    if (lane == 0) { cr = 0.0f; ci = 0.0f; }
    float sr = cr, si = ci;
    for (int t = 0; t < 128; ++t) {
        size_t row = base + (size_t)(l0 + t) * Nd;
        float br = bu[row + p];
        float bi = bu[row + p + Pd];
        float nr = lre * sr - lim * si + br;
        float ni = lre * si + lim * sr + bi;
        sr = nr; si = ni;
        bu[row + p]      = sr;
        bu[row + p + Pd] = si;
    }
}

// ---------------------------------------------------------------------------
// Kernel 6: GEMM2  y[m][h] = sum_k xs[m][k]*W2[k][h];  y = gelu(y + D[h]*u);
// output transposed through LDS so stores are l-contiguous in (b,h,l) layout.
// ---------------------------------------------------------------------------
__global__ void s5_gemm2(const float* __restrict__ in,
                         const __bf16* __restrict__ W2,
                         const float* __restrict__ xs,
                         const float* __restrict__ Dv,
                         float* __restrict__ out) {
    __shared__ __bf16 At[64][AROW];
    __shared__ float  Ct[8][16][17];    // padded transpose tile per wave
    int tid  = threadIdx.x;
    int wave = tid >> 5;
    int lane = tid & 31;
    int m0   = blockIdx.x * 64;
    int n0   = blockIdx.y * 128 + wave * 16;
    int ntile = n0 >> 4;
    int b  = m0 >> 12;
    int l0 = m0 & (Ld - 1);

    // stage A panel: xs is row-major -> float2 loads, fully coalesced rows
#pragma unroll 4
    for (int m = 0; m < 64; ++m) {
        int k2 = tid;                   // 0..255 pair index
        const float2* row =
            (const float2*)(xs + (size_t)(m0 + m) * Nd);
        float2 v = row[k2];
        *(unsigned*)&At[m][2 * k2] = pack_bf16x2(v.x, v.y);
    }
    __syncthreads();

    int hi   = lane >> 4;
    int mrow = lane & 15;
    v8f acc[4] = {};
    for (int kt = 0; kt < Kd / 32; ++kt) {
        const v8bf* wp = (const v8bf*)(W2 + ((size_t)(kt * 32 + ntile)) * 512
                                          + lane * 16);
        v16bf bfrag = cat8(wp[0], wp[1]);
#pragma unroll
        for (int sub = 0; sub < 4; ++sub) {
            const v8bf* arow = (const v8bf*)(&At[sub * 16 + mrow][kt * 32]);
            v16bf afrag = cat8(arow[hi], arow[hi + 2]);
            acc[sub] = __builtin_amdgcn_wmma_f32_16x16x32_bf16(
                false, afrag, false, bfrag, (short)0, acc[sub], false, false);
        }
    }

    // epilogue per 16x16 subtile: transpose via LDS, fuse D*u + exact GeLU
    int n  = lane >> 1;                 // h column within tile
    int mb = (lane & 1) * 8;            // l half
    int h  = n0 + n;
    float d = Dv[h];
#pragma unroll
    for (int sub = 0; sub < 4; ++sub) {
        __syncthreads();
#pragma unroll
        for (int r = 0; r < 8; ++r)
            Ct[wave][r + 8 * hi][lane & 15] = acc[sub][r];
        __syncthreads();
        size_t col = (size_t)b * Hd * Ld + (size_t)h * Ld
                   + (l0 + sub * 16 + mb);
        const float* ucol = in + col;
        float* ocol = out + col;
#pragma unroll
        for (int j = 0; j < 8; ++j) {
            float y = Ct[wave][mb + j][n] + d * ucol[j];
            y = 0.5f * y * (1.0f + erff(y * 0.70710678118654752f));
            ocol[j] = y;
        }
    }
}

// ---------------------------------------------------------------------------
extern "C" void kernel_launch(void* const* d_in, const int* in_sizes, int n_in,
                              void* d_out, int out_size, void* d_ws, size_t ws_size,
                              hipStream_t stream) {
    const float* in_seq   = (const float*)d_in[0];
    const float* Lre      = (const float*)d_in[1];
    const float* Lim      = (const float*)d_in[2];
    const float* Bmat     = (const float*)d_in[3];
    const float* Cmat     = (const float*)d_in[4];
    const float* Dvec     = (const float*)d_in[5];
    const float* log_step = (const float*)d_in[6];
    float* out = (float*)d_out;

    // workspace layout
    char* ws = (char*)d_ws;
    float*  bu  = (float*)ws;                                   // 64 MB Bu/xs
    size_t  off = (size_t)Md * Nd * sizeof(float);
    float*  lam = (float*)(ws + off);                           // 512 f32
    float*  g   = lam + 2 * Pd;                                 // 512 f32
    __bf16* W1  = (__bf16*)(ws + off + 8192);                   // 512 KB
    __bf16* W2  = W1 + (size_t)Kd * Nd;                         // 512 KB

    s5_discretize<<<1, 256, 0, stream>>>(Lre, Lim, log_step, lam, g);
    s5_pack_w1<<<64, 256, 0, stream>>>(Bmat, g, W1);
    s5_pack_w2<<<64, 256, 0, stream>>>(Cmat, W2);

    dim3 ggrid(Md / 64, Nd / 128);
    s5_gemm1<<<ggrid, 256, 0, stream>>>(in_seq, W1, bu);
    s5_scan<<<dim3(Pd / 8, BSZ), 256, 0, stream>>>(lam, bu);
    s5_gemm2<<<ggrid, 256, 0, stream>>>(in_seq, W2, bu, Dvec, out);
}
